// EquivariantAttention_42374147342461
// MI455X (gfx1250) — matrix operand order repaired and checked by reading, not compiled, not run on hardware
//
#include <hip/hip_runtime.h>
#include <hip/hip_bf16.h>

typedef __attribute__((ext_vector_type(16))) _Float16 v16h;
typedef __attribute__((ext_vector_type(8)))  float    v8f;

#define NN 512

union AFrag { v16h v; float4 f4[2]; };

__device__ __forceinline__ v8f splat8(float x) {
    v8f c;
#pragma unroll
    for (int i = 0; i < 8; ++i) c[i] = x;
    return c;
}

#define WMMA(A, B, C) __builtin_amdgcn_wmma_f32_16x16x32_f16(false, (A), false, (B), (short)0, (C), false, false)
#define WAIT_DS() asm volatile("s_wait_dscnt 0" ::: "memory")

__device__ __forceinline__ float fast_tanh(float x) {
    // tanh(x) = 1 - 2/(exp(2x)+1);  exp(2x) = exp2(x * 2/ln2)
    float e = __builtin_amdgcn_exp2f(x * 2.8853900817779268f);
    return 1.0f - 2.0f * __builtin_amdgcn_rcpf(e + 1.0f);
}

// ---------------------------------------------------------------------------
// Kernel 1: per-node precompute.
//  q/k/v (e3_linear + head split, CG*ATTN_NORM folded into q), padded to 64.
//  A_recT / A_sndT [n][o(64)][c(32, 9 valid)] f16 (CG folded) for WMMA B-frags.
//  Block 512 transposes MLP weights into WMMA-B-friendly f16 layouts.
// ---------------------------------------------------------------------------
__global__ __launch_bounds__(256) void k1_prep(
    const float* __restrict__ nf_g,
    const float* __restrict__ Wq0, const float* __restrict__ Wq1, const float* __restrict__ Wq2,
    const float* __restrict__ Wk0, const float* __restrict__ Wk1, const float* __restrict__ Wk2,
    const float* __restrict__ Wv0, const float* __restrict__ Wv1, const float* __restrict__ Wv2,
    const float* __restrict__ mW1, const float* __restrict__ mW2, const float* __restrict__ mW3,
    _Float16* __restrict__ Q, _Float16* __restrict__ Kh, _Float16* __restrict__ V,
    _Float16* __restrict__ AR, _Float16* __restrict__ AS,
    _Float16* __restrict__ W1aT, _Float16* __restrict__ W2T, _Float16* __restrict__ W3T)
{
    int n = blockIdx.x, t = threadIdx.x;
    if (n == NN) {  // weight transposes
        for (int i = t; i < 2048; i += 256) { int o = i >> 5, c = i & 31; W1aT[o * 32 + c] = (_Float16)mW1[c * 64 + o]; }
        for (int i = t; i < 4096; i += 256) { int o = i >> 6, k = i & 63; W2T[o * 64 + k] = (_Float16)mW2[k * 64 + o]; }
        for (int i = t; i < 1024; i += 256) { int o = i >> 6, k = i & 63; W3T[o * 64 + k] = (_Float16)((o < 8) ? mW3[k * 8 + o] : 0.0f); }
        return;
    }
    __shared__ float snf[480];
    for (int i = t; i < 480; i += 256) snf[i] = nf_g[(size_t)n * 480 + i];
    __syncthreads();

    const float* Wt[9] = {Wq0, Wq1, Wq2, Wk0, Wk1, Wk2, Wv0, Wv1, Wv2};
    const float CG[3] = {1.0f, 0.5773502691896258f, 0.4472135954999579f};  // 1/sqrt(2l+1)
    const float ATT = 0.1889822365046136f;                                 // 1/sqrt(28)

    // q, k, v
    for (int idx = t; idx < 3 * 480; idx += 256) {
        int w = idx / 480, rem = idx - w * 480;
        int l, o, m, mul, d, base, am, jb;
        if (rem < 128)      { l = 0; o = rem;            m = 0;                 mul = 128; d = 1; base = 0;   am = 16; jb = 0;  }
        else if (rem < 320) { int u = rem - 128; l = 1; o = u / 3; m = u % 3;  mul = 64;  d = 3; base = 128; am = 8;  jb = 16; }
        else                { int u = rem - 320; l = 2; o = u / 5; m = u % 5;  mul = 32;  d = 5; base = 320; am = 4;  jb = 40; }
        const float* W = Wt[w * 3 + l];
        float s = 0.0f;
        for (int i = 0; i < mul; ++i) s += snf[base + i * d + m] * W[i * mul + o];
        s *= rsqrtf((float)mul);
        int h = o / am, i2 = o - h * am, j = jb + i2 * d + m;
        if (w == 0)      { s *= CG[l] * ATT; Q[((size_t)n * 8 + h) * 64 + j] = (_Float16)s; }
        else if (w == 1) { Kh[((size_t)n * 8 + h) * 64 + j] = (_Float16)s; }
        else             { V[((size_t)(h * 64 + j)) * NN + n] = (_Float16)s; }
    }
    // A_rec / A_snd (CG folded)
    for (int idx = t; idx < 1152; idx += 256) {
        int w2 = idx / 576, rem = idx - w2 * 576, c = rem >> 6, o = rem & 63;
        int l, m, rb, mul, d, base;
        if (c == 0)     { l = 0; m = 0;     rb = 0;   mul = 128; d = 1; base = 0;   }
        else if (c < 4) { l = 1; m = c - 1; rb = 128; mul = 64;  d = 3; base = 128; }
        else            { l = 2; m = c - 4; rb = 192; mul = 32;  d = 5; base = 320; }
        const float* Wrow = mW1 + (size_t)(32 + w2 * 224 + rb) * 64 + o;
        float s = 0.0f;
        for (int i = 0; i < mul; ++i) s += snf[base + i * d + m] * Wrow[(size_t)i * 64];
        s *= CG[l];
        (w2 ? AS : AR)[((size_t)n * 64 + o) * 32 + c] = (_Float16)s;
    }
    // zero pads: q/k j=60..63, vT j=60..63, A*T c=9..31
    for (int idx = t; idx < 64; idx += 256) {
        int w = idx >> 5, rem = idx & 31, h = rem >> 2, j = 60 + (rem & 3);
        (w ? Kh : Q)[((size_t)n * 8 + h) * 64 + j] = (_Float16)0.0f;
    }
    for (int idx = t; idx < 32; idx += 256) {
        int h = idx >> 2, j = 60 + (idx & 3);
        V[((size_t)(h * 64 + j)) * NN + n] = (_Float16)0.0f;
    }
    for (int idx = t; idx < 2944; idx += 256) {
        int w = idx / 1472, rem = idx - w * 1472, o = rem / 23, c = 9 + rem % 23;
        (w ? AS : AR)[((size_t)n * 64 + o) * 32 + c] = (_Float16)0.0f;
    }
}

// ---------------------------------------------------------------------------
// Kernel 2a: sender-major term:  partial[r][s][o] = sum_c sh[r,s,c]*A_snd[s,c,o]
// Wave = fixed s; B-frags (A_sndT[s]) resident; loop over 16-receiver tiles.
// ---------------------------------------------------------------------------
__global__ __launch_bounds__(256) void k2a_partial(
    const float* __restrict__ sh_g, const _Float16* __restrict__ AS, _Float16* __restrict__ P)
{
    __shared__ _Float16 tile[8][16][64];
    int wave = threadIdx.x >> 5, lane = threadIdx.x & 31, cl = lane & 15, hi = lane >> 4;
    int sg = blockIdx.x >> 2, rc = blockIdx.x & 3;
    int s = sg * 8 + wave;

    v16h bfr[4];
#pragma unroll
    for (int nt = 0; nt < 4; ++nt)
        bfr[nt] = *(const v16h*)(AS + ((size_t)s * 64 + nt * 16 + cl) * 32 + hi * 16);

    for (int rt = 0; rt < 8; ++rt) {
        int r0 = (rc * 8 + rt) * 16;
        AFrag a;
#pragma unroll
        for (int e = 0; e < 16; ++e) a.v[e] = (_Float16)0.0f;
        const float* shp = sh_g + (((size_t)(r0 + cl)) * NN + s) * 9;
        if (lane < 16) {
#pragma unroll
            for (int c = 0; c < 9; ++c) a.v[c] = (_Float16)shp[c];
        }
        v8f acc[4];
#pragma unroll
        for (int nt = 0; nt < 4; ++nt) acc[nt] = WMMA(a.v, bfr[nt], splat8(0.0f));
#pragma unroll
        for (int nt = 0; nt < 4; ++nt)
#pragma unroll
            for (int v = 0; v < 8; ++v) tile[wave][v + 8 * hi][nt * 16 + cl] = (_Float16)acc[nt][v];
        WAIT_DS();
        int rr = lane >> 1, hh = lane & 1;
        const float4* src = (const float4*)&tile[wave][rr][hh * 32];
        float4* dst = (float4*)(P + (((size_t)(r0 + rr)) * NN + s) * 64 + hh * 32);
#pragma unroll
        for (int q4 = 0; q4 < 4; ++q4) dst[q4] = src[q4];
        // DS ops are in-order within a wave: next iteration's stores can't pass these loads.
    }
}

// ---------------------------------------------------------------------------
// Kernel 2b: receiver-major fused MLP.
//  h1 = tanh(ea@W1a + sh@A_rec[r] + partial + b1); h2 = tanh(h1@W2+b2); ew = h2@W3+b3
//  LDS bounce converts C-layout f32 -> A-layout f16 between layers.
// ---------------------------------------------------------------------------
__global__ __launch_bounds__(256) void k2b_mlp(
    const float* __restrict__ ea_g, const float* __restrict__ sh_g,
    const float* __restrict__ mb1, const float* __restrict__ mb2, const float* __restrict__ mb3,
    const _Float16* __restrict__ W1aT, const _Float16* __restrict__ W2T, const _Float16* __restrict__ W3T,
    const _Float16* __restrict__ AR, const _Float16* __restrict__ P, float* __restrict__ EW)
{
    __shared__ _Float16 tile[8][16][64];
    int wave = threadIdx.x >> 5, lane = threadIdx.x & 31, cl = lane & 15, hi = lane >> 4;
    int rg = blockIdx.x >> 2, sc = blockIdx.x & 3;
    int r = rg * 8 + wave;
    int koff = hi * 16;

    v16h fW1[4], fAr[4], fW2[4][2], fW3[2];
#pragma unroll
    for (int nt = 0; nt < 4; ++nt) {
        fW1[nt] = *(const v16h*)(W1aT + (nt * 16 + cl) * 32 + hi * 16);
        fAr[nt] = *(const v16h*)(AR + ((size_t)r * 64 + nt * 16 + cl) * 32 + hi * 16);
#pragma unroll
        for (int ks = 0; ks < 2; ++ks)
            fW2[nt][ks] = *(const v16h*)(W2T + (nt * 16 + cl) * 64 + ks * 32 + hi * 16);
    }
#pragma unroll
    for (int ks = 0; ks < 2; ++ks) fW3[ks] = *(const v16h*)(W3T + cl * 64 + ks * 32 + hi * 16);
    float b1v[4], b2v[4];
#pragma unroll
    for (int nt = 0; nt < 4; ++nt) { b1v[nt] = mb1[nt * 16 + cl]; b2v[nt] = mb2[nt * 16 + cl]; }
    float b3v = (cl < 8) ? mb3[cl] : 0.0f;

    for (int st = 0; st < 8; ++st) {
        int s0 = (sc * 8 + st) * 16;
        // --- edge_attr A-frag (f32 -> f16) ---
        AFrag aea;
        const float* eap = ea_g + (((size_t)r * NN) + s0 + cl) * 32;
        float4 p0 = *(const float4*)(eap + koff);
        float4 p1 = *(const float4*)(eap + koff + 4);
        float4 p2 = *(const float4*)(eap + 8 + koff);
        float4 p3 = *(const float4*)(eap + 12 + koff);
        aea.v[0] = (_Float16)p0.x;  aea.v[1] = (_Float16)p0.y;  aea.v[2]  = (_Float16)p0.z;  aea.v[3]  = (_Float16)p0.w;
        aea.v[4] = (_Float16)p1.x;  aea.v[5] = (_Float16)p1.y;  aea.v[6]  = (_Float16)p1.z;  aea.v[7]  = (_Float16)p1.w;
        aea.v[8] = (_Float16)p2.x;  aea.v[9] = (_Float16)p2.y;  aea.v[10] = (_Float16)p2.z;  aea.v[11] = (_Float16)p2.w;
        aea.v[12] = (_Float16)p3.x; aea.v[13] = (_Float16)p3.y; aea.v[14] = (_Float16)p3.z;  aea.v[15] = (_Float16)p3.w;
        // --- sh A-frag (K padded 9 -> 32) ---
        AFrag ash;
#pragma unroll
        for (int e = 0; e < 16; ++e) ash.v[e] = (_Float16)0.0f;
        const float* shp = sh_g + (((size_t)r * NN) + s0 + cl) * 9;
        if (lane < 16) {
#pragma unroll
            for (int c = 0; c < 9; ++c) ash.v[c] = (_Float16)shp[c];
        }
        // --- layer 1 (bias via C input) ---
        v8f acc1[4];
#pragma unroll
        for (int nt = 0; nt < 4; ++nt) {
            v8f c = WMMA(aea.v, fW1[nt], splat8(b1v[nt]));
            acc1[nt] = WMMA(ash.v, fAr[nt], c);
        }
        // --- stage partial (sender term) tile into LDS ---
        {
            int rr = lane >> 1, hh = lane & 1;
            const float4* gp = (const float4*)(P + (((size_t)r * NN) + s0 + rr) * 64 + hh * 32);
            float4* lp = (float4*)&tile[wave][rr][hh * 32];
#pragma unroll
            for (int q4 = 0; q4 < 4; ++q4) lp[q4] = gp[q4];
        }
        WAIT_DS();
        // --- add partial, tanh, write f16 back (same slots, same lane: DS in-order) ---
#pragma unroll
        for (int nt = 0; nt < 4; ++nt)
#pragma unroll
            for (int v = 0; v < 8; ++v) {
                float x = acc1[nt][v] + (float)tile[wave][v + 8 * hi][nt * 16 + cl];
                tile[wave][v + 8 * hi][nt * 16 + cl] = (_Float16)fast_tanh(x);
            }
        WAIT_DS();
        // --- layer 2 ---
        AFrag a2[2];
#pragma unroll
        for (int ks = 0; ks < 2; ++ks) {
            const char* base = (const char*)&tile[wave][cl][0];
            a2[ks].f4[0] = *(const float4*)(base + ks * 64 + koff * 2);
            a2[ks].f4[1] = *(const float4*)(base + ks * 64 + koff * 2 + 16);
        }
        v8f acc2[4];
#pragma unroll
        for (int nt = 0; nt < 4; ++nt) {
            v8f c = WMMA(a2[0].v, fW2[nt][0], splat8(b2v[nt]));
            acc2[nt] = WMMA(a2[1].v, fW2[nt][1], c);
        }
#pragma unroll
        for (int nt = 0; nt < 4; ++nt)
#pragma unroll
            for (int v = 0; v < 8; ++v)
                tile[wave][v + 8 * hi][nt * 16 + cl] = (_Float16)fast_tanh(acc2[nt][v]);
        WAIT_DS();
        // --- layer 3 (N padded to 16; cols 0..7 valid = heads) ---
        AFrag a3[2];
#pragma unroll
        for (int ks = 0; ks < 2; ++ks) {
            const char* base = (const char*)&tile[wave][cl][0];
            a3[ks].f4[0] = *(const float4*)(base + ks * 64 + koff * 2);
            a3[ks].f4[1] = *(const float4*)(base + ks * 64 + koff * 2 + 16);
        }
        v8f c3 = WMMA(a3[0].v, fW3[0], splat8(b3v));
        c3 = WMMA(a3[1].v, fW3[1], c3);
        // --- store edge_weight [h][r][s] (C rows are consecutive s) ---
        if (cl < 8) {
            float* op = EW + (((size_t)cl * NN) + r) * NN + s0 + hi * 8;
            float4 lo, hh4;
            lo.x = c3[0]; lo.y = c3[1]; lo.z = c3[2]; lo.w = c3[3];
            hh4.x = c3[4]; hh4.y = c3[5]; hh4.z = c3[6]; hh4.w = c3[7];
            *(float4*)op = lo;
            *(float4*)(op + 4) = hh4;
        }
    }
}

// ---------------------------------------------------------------------------
// Kernel 3: qk logits per head: LQK[h][r][s] = sum_j q[r,h,j]*k[s,h,j]
// A = K rows (s), B = Q^T (col = r) -> C rows are consecutive s => coalesced store.
// ---------------------------------------------------------------------------
__global__ __launch_bounds__(256) void k3_qk(
    const _Float16* __restrict__ Q, const _Float16* __restrict__ Kh, float* __restrict__ LQK)
{
    int wave = threadIdx.x >> 5, lane = threadIdx.x & 31, cl = lane & 15, hi = lane >> 4;
    int gid = blockIdx.x * 8 + wave;
    int h = gid >> 10, rem = gid & 1023, rt = rem >> 5, stt = rem & 31;
    int r0 = rt * 16, s0 = stt * 16, koff = hi * 16;

    AFrag a[2]; v16h b[2];
#pragma unroll
    for (int ks = 0; ks < 2; ++ks) {
        const _Float16* kp = Kh + (((size_t)(s0 + cl)) * 8 + h) * 64 + ks * 32;
        a[ks].f4[0] = *(const float4*)(kp + koff);
        a[ks].f4[1] = *(const float4*)(kp + 8 + koff);
        b[ks] = *(const v16h*)(Q + (((size_t)(r0 + cl)) * 8 + h) * 64 + ks * 32 + hi * 16);
    }
    v8f c = WMMA(a[0].v, b[0], splat8(0.0f));
    c = WMMA(a[1].v, b[1], c);
    float* op = LQK + (((size_t)h * NN) + (r0 + cl)) * NN + s0 + hi * 8;
    float4 lo, hh4;
    lo.x = c[0]; lo.y = c[1]; lo.z = c[2]; lo.w = c[3];
    hh4.x = c[4]; hh4.y = c[5]; hh4.z = c[6]; hh4.w = c[7];
    *(float4*)op = lo;
    *(float4*)(op + 4) = hh4;
}

// ---------------------------------------------------------------------------
// Kernel 4: softmax over senders for each (r,h); writes f16 weights [h][r][s].
// ---------------------------------------------------------------------------
__global__ __launch_bounds__(256) void k4_softmax(
    const float* __restrict__ EW, const float* __restrict__ LQK, _Float16* __restrict__ WG)
{
    int r = blockIdx.x, h = threadIdx.x >> 5, lane = threadIdx.x & 31;
    size_t base = ((size_t)h * NN + r) * NN;
    float x[16], m = -1e30f;
#pragma unroll
    for (int t = 0; t < 16; ++t) {
        int s = lane + t * 32;
        x[t] = EW[base + s] + LQK[base + s];
        m = fmaxf(m, x[t]);
    }
#pragma unroll
    for (int d = 16; d >= 1; d >>= 1) m = fmaxf(m, __shfl_xor(m, d, 32));
    float e[16], sum = 0.0f;
#pragma unroll
    for (int t = 0; t < 16; ++t) {
        e[t] = __builtin_amdgcn_exp2f((x[t] - m) * 1.4426950408889634f);
        sum += e[t];
    }
#pragma unroll
    for (int d = 16; d >= 1; d >>= 1) sum += __shfl_xor(sum, d, 32);
    float inv = __builtin_amdgcn_rcpf(sum);
#pragma unroll
    for (int t = 0; t < 16; ++t) WG[base + lane + t * 32] = (_Float16)(e[t] * inv);
}

// ---------------------------------------------------------------------------
// Kernel 5: aggregation  upd[r,h,j] = sum_s W[h,r,s]*vT[h,j,s]  (K=512, 16 WMMA)
// then residual add + scatter into irreps layout.
// ---------------------------------------------------------------------------
__global__ __launch_bounds__(256) void k5_agg(
    const _Float16* __restrict__ WG, const _Float16* __restrict__ V,
    const float* __restrict__ nf_g, float* __restrict__ out)
{
    int wave = threadIdx.x >> 5, lane = threadIdx.x & 31, cl = lane & 15, hi = lane >> 4;
    int gid = blockIdx.x * 8 + wave;
    int h = gid >> 7, rem = gid & 127, rt = rem >> 2, jt = rem & 3;
    int r0 = rt * 16, j0 = jt * 16, koff = hi * 16;

    v8f acc = splat8(0.0f);
    for (int ks = 0; ks < 16; ++ks) {
        AFrag a;
        const _Float16* wp = WG + (((size_t)h * NN) + (r0 + cl)) * NN + ks * 32;
        a.f4[0] = *(const float4*)(wp + koff);
        a.f4[1] = *(const float4*)(wp + 8 + koff);
        v16h b = *(const v16h*)(V + (((size_t)h * 64) + (j0 + cl)) * NN + ks * 32 + hi * 16);
        acc = WMMA(a.v, b, acc);
    }
    int j = j0 + cl, col;
    if (j < 16)      col = h * 16 + j;
    else if (j < 40) { int u = j - 16; col = 128 + (h * 8 + u / 3) * 3 + u % 3; }
    else if (j < 60) { int u = j - 40; col = 320 + (h * 4 + u / 5) * 5 + u % 5; }
    else             col = -1;
    if (col >= 0) {
#pragma unroll
        for (int v = 0; v < 8; ++v) {
            int rr = r0 + v + 8 * hi;
            out[(size_t)rr * 480 + col] = nf_g[(size_t)rr * 480 + col] + acc[v];
        }
    }
}

// ---------------------------------------------------------------------------
extern "C" void kernel_launch(void* const* d_in, const int* in_sizes, int n_in,
                              void* d_out, int out_size, void* d_ws, size_t ws_size,
                              hipStream_t stream) {
    (void)in_sizes; (void)n_in; (void)out_size; (void)ws_size;
    const float* node_feat = (const float*)d_in[0];
    const float* edge_attr = (const float*)d_in[1];
    const float* edge_sh   = (const float*)d_in[2];
    const float* Wq0 = (const float*)d_in[3],  *Wq1 = (const float*)d_in[4],  *Wq2 = (const float*)d_in[5];
    const float* Wk0 = (const float*)d_in[6],  *Wk1 = (const float*)d_in[7],  *Wk2 = (const float*)d_in[8];
    const float* Wv0 = (const float*)d_in[9],  *Wv1 = (const float*)d_in[10], *Wv2 = (const float*)d_in[11];
    const float* mW1 = (const float*)d_in[12], *mb1 = (const float*)d_in[13];
    const float* mW2 = (const float*)d_in[14], *mb2 = (const float*)d_in[15];
    const float* mW3 = (const float*)d_in[16], *mb3 = (const float*)d_in[17];

    char* ws = (char*)d_ws;
    size_t off = 0;
    auto alloc = [&](size_t bytes) { void* p = ws + off; off += (bytes + 255) & ~(size_t)255; return p; };
    _Float16* Q    = (_Float16*)alloc((size_t)NN * 8 * 64 * 2);       // 512 KB
    _Float16* Kh   = (_Float16*)alloc((size_t)NN * 8 * 64 * 2);       // 512 KB
    _Float16* V    = (_Float16*)alloc((size_t)8 * 64 * NN * 2);       // 512 KB
    _Float16* AR   = (_Float16*)alloc((size_t)NN * 64 * 32 * 2);      // 2 MB
    _Float16* AS   = (_Float16*)alloc((size_t)NN * 64 * 32 * 2);      // 2 MB
    _Float16* W1aT = (_Float16*)alloc(2048 * 2);
    _Float16* W2T  = (_Float16*)alloc(4096 * 2);
    _Float16* W3T  = (_Float16*)alloc(1024 * 2);
    _Float16* P    = (_Float16*)alloc((size_t)NN * NN * 64 * 2);      // 32 MB
    float*    EW   = (float*)   alloc((size_t)NN * NN * 8 * 4);       // 8 MB
    float*    LQK  = (float*)   alloc((size_t)NN * NN * 8 * 4);       // 8 MB
    _Float16* WG   = (_Float16*)alloc((size_t)NN * NN * 8 * 2);       // 4 MB

    k1_prep<<<513, 256, 0, stream>>>(node_feat, Wq0, Wq1, Wq2, Wk0, Wk1, Wk2, Wv0, Wv1, Wv2,
                                     mW1, mW2, mW3, Q, Kh, V, AR, AS, W1aT, W2T, W3T);
    k2a_partial<<<256, 256, 0, stream>>>(edge_sh, AS, P);
    k2b_mlp<<<256, 256, 0, stream>>>(edge_attr, edge_sh, mb1, mb2, mb3, W1aT, W2T, W3T, AR, P, EW);
    k3_qk<<<1024, 256, 0, stream>>>(Q, Kh, LQK);
    k4_softmax<<<512, 256, 0, stream>>>(EW, LQK, WG);
    k5_agg<<<128, 256, 0, stream>>>(WG, V, node_feat, (float*)d_out);
}
